// Self_Attention_3315714752746
// MI455X (gfx1250) — matrix operand order repaired and checked by reading, not compiled, not run on hardware
//
#include <hip/hip_runtime.h>
#include <hip/hip_bf16.h>

typedef __bf16 bf16_t;
typedef __attribute__((ext_vector_type(16))) __bf16 v16bf;
typedef __attribute__((ext_vector_type(8)))  float  v8f;
typedef unsigned short ushort_t;
typedef unsigned int   uint32_t_;

#define NB 4
#define CC 256
#define NN 4096
#define C8V 32

// round-to-nearest-even float -> bf16 bits
__device__ __forceinline__ ushort_t f2bf(float f) {
  union { float f; unsigned int u; } v; v.f = f;
  unsigned int r = (v.u + 0x7FFFu + ((v.u >> 16) & 1u)) >> 16;
  return (ushort_t)r;
}

__device__ __forceinline__ v8f wmma_bf16(v16bf a, v16bf b, v8f c) {
  // v_wmma_f32_16x16x32_bf16  D = A(16x32) * B(32x16) + C
  return __builtin_amdgcn_wmma_f32_16x16x32_bf16(
      /*neg_a=*/false, a, /*neg_b=*/false, b,
      /*c_mod=*/(short)0, c, /*reuse_a=*/false, /*reuse_b=*/false);
}

// ---------------------------------------------------------------------------
// Kernel 1: q,k projections (K=256 dot per output), stored pre-swizzled bf16.
// qA fragment (A-matrix layout, 16x32): lane=(n&15)+16*half, elem e -> K_A
//   K_A(e,half) = (e<8 ? e : e+8) + 8*half   (ISA 16-bit A table)
// kB fragment (B-matrix layout, 32x16): lane=(m&15)+16*half, elem e -> K_B
//   K_B(e,half) = (e<8 ? e : e+8) + 8*half   (same k order as A)
// ---------------------------------------------------------------------------
__global__ __launch_bounds__(256) void proj_qk_kernel(
    const float* __restrict__ x,  const float* __restrict__ Wq,
    const float* __restrict__ bq, const float* __restrict__ Wk,
    const float* __restrict__ bk,
    ushort_t* __restrict__ qA, ushort_t* __restrict__ kB)
{
  const int b = blockIdx.y;
  const int n = blockIdx.x * 256 + threadIdx.x;         // spatial position
  const float* xp = x + (size_t)b * CC * NN + n;

  float qacc[C8V], kacc[C8V];
#pragma unroll
  for (int o = 0; o < C8V; ++o) { qacc[o] = 0.f; kacc[o] = 0.f; }

  for (int c = 0; c < CC; ++c) {
    const float xv = xp[(size_t)c * NN];
#pragma unroll
    for (int o = 0; o < C8V; ++o) {
      qacc[o] = fmaf(Wq[o * CC + c], xv, qacc[o]);
      kacc[o] = fmaf(Wk[o * CC + c], xv, kacc[o]);
    }
  }
#pragma unroll
  for (int o = 0; o < C8V; ++o) { qacc[o] += bq[o]; kacc[o] += bk[o]; }

  const int nt = n >> 4;
#pragma unroll
  for (int half = 0; half < 2; ++half) {
    const int lane = (n & 15) + 16 * half;
    const size_t base = (((size_t)b * 256 + nt) * 32 + lane) * 16;
#pragma unroll
    for (int e = 0; e < 16; ++e) {
      const int Ka = (e < 8 ? e : e + 8) + 8 * half;   // A layout (q rows)
      const int Kb = (e < 8 ? e : e + 8) + 8 * half;   // B layout (k cols)
      qA[base + e] = f2bf(qacc[Ka]);
      kB[base + e] = f2bf(kacc[Kb]);
    }
  }
}

// ---------------------------------------------------------------------------
// Kernel 2: pv projection (C outputs per n), stored pre-swizzled bf16 as
// B-fragments for the final GEMM: pvB[b][kt(128)][ot(16)][lane(32)][e(16)],
// value = pv[o = ot*16+(lane&15)][n = kt*32 + K_B(e,half)].
// ---------------------------------------------------------------------------
__global__ __launch_bounds__(256) void proj_v_kernel(
    const float* __restrict__ x, const float* __restrict__ Wv,
    const float* __restrict__ bv, ushort_t* __restrict__ pvB)
{
  const int b  = blockIdx.y;
  const int n0 = blockIdx.x * 16;
  const int tid = threadIdx.x;
  __shared__ float sx[CC * 16];                // x[c][n0..n0+15], 16 KB

#pragma unroll
  for (int i = 0; i < 16; ++i) {
    const int idx = tid + i * 256;             // idx = c*16 + nl
    const int c = idx >> 4, nl = idx & 15;
    sx[idx] = x[((size_t)b * CC + c) * NN + n0 + nl];
  }
  __syncthreads();

  const int o = tid;                           // one output channel per thread
  float acc[16];
#pragma unroll
  for (int j = 0; j < 16; ++j) acc[j] = 0.f;
  const float* wrow = Wv + (size_t)o * CC;
  for (int c = 0; c < CC; ++c) {
    const float w = wrow[c];
#pragma unroll
    for (int j = 0; j < 16; ++j) acc[j] = fmaf(w, sx[c * 16 + j], acc[j]);
  }
  const float bias = bv[o];
  const int kt = n0 >> 5;
  const int ot = o >> 4;
#pragma unroll
  for (int j = 0; j < 16; ++j) {
    const int K    = (n0 & 31) + j;            // K index within the 32-tile
    const int half = (K >> 3) & 1;
    const int e    = (K & 7) + ((K >> 4) << 3);
    const int lane = (o & 15) + 16 * half;
    const size_t idx = ((((size_t)b * 128 + kt) * 16 + ot) * 32 + lane) * 16 + e;
    pvB[idx] = f2bf(acc[j] + bias);
  }
}

// ---------------------------------------------------------------------------
// Kernel 3: energy GEMM + softmax + column mask + row renorm -> sa (fp32).
// Block = (b, 16-query-row tile). 8 waves split 4096 cols (512 each).
// Two passes over m: pass1 accumulates row sums of exp(e)*valid in registers
// (ds_add_f32 reduce), pass2 recomputes and writes normalized sa.
// ---------------------------------------------------------------------------
__global__ __launch_bounds__(256) void attn_softmax_kernel(
    const ushort_t* __restrict__ qA, const ushort_t* __restrict__ kB,
    const float* __restrict__ valid, float* __restrict__ sa)
{
  const int b    = blockIdx.y;
  const int nt   = blockIdx.x;                 // query-row tile (16 rows)
  const int tid  = threadIdx.x;
  const int wave = tid >> 5;
  const int lane = tid & 31;
  const int l15  = lane & 15;
  const int half = lane >> 4;

  __shared__ float rowsum[16];
  if (tid < 16) rowsum[tid] = 0.f;

  const v16bf afrag =
      *(const v16bf*)(qA + (((size_t)b * 256 + nt) * 32 + lane) * 16);
  const size_t kbase = (size_t)b * 256 * 32 * 16;

  float partial[8];
#pragma unroll
  for (int r = 0; r < 8; ++r) partial[r] = 0.f;

  for (int j = 0; j < 32; ++j) {               // pass 1: sums only
    const int mt = wave * 32 + j;
    const v16bf bfrag = *(const v16bf*)(kB + kbase + ((size_t)mt * 32 + lane) * 16);
    v8f acc = {0.f,0.f,0.f,0.f,0.f,0.f,0.f,0.f};
    acc = wmma_bf16(afrag, bfrag, acc);
    const int m = mt * 16 + l15;
    const float v = valid[(size_t)b * NN + m];
#pragma unroll
    for (int r = 0; r < 8; ++r) partial[r] += __expf(acc[r]) * v;
  }
  __syncthreads();                             // rowsum init visible
#pragma unroll
  for (int r = 0; r < 8; ++r) atomicAdd(&rowsum[r + 8 * half], partial[r]);
  __syncthreads();

  float inv[8];
#pragma unroll
  for (int r = 0; r < 8; ++r) inv[r] = 1.0f / rowsum[r + 8 * half];

  float* sarow0 = sa + ((size_t)b * NN + (size_t)nt * 16) * NN;
  for (int j = 0; j < 32; ++j) {               // pass 2: write normalized sa
    const int mt = wave * 32 + j;
    const v16bf bfrag = *(const v16bf*)(kB + kbase + ((size_t)mt * 32 + lane) * 16);
    v8f acc = {0.f,0.f,0.f,0.f,0.f,0.f,0.f,0.f};
    acc = wmma_bf16(afrag, bfrag, acc);
    const int m = mt * 16 + l15;
    const float v = valid[(size_t)b * NN + m];
#pragma unroll
    for (int r = 0; r < 8; ++r) {
      const int row = r + 8 * half;
      sarow0[(size_t)row * NN + m] = __expf(acc[r]) * v * inv[r];
    }
  }
}

// ---------------------------------------------------------------------------
// Kernel 4: out[b,o,m] = gamma * (sum_n pv[o,n]*sa[m,n]) + x[b,o,m].
// Block tile 64(m) x 64(o), 8 waves as 4x2; wave = 16m x 32o (2 accumulators).
// A = sa rows (fp32 from d_out, converted to bf16 in regs), B = pvB fragments.
// ---------------------------------------------------------------------------
__global__ __launch_bounds__(256) void attn_out_kernel(
    const float* __restrict__ sa, const ushort_t* __restrict__ pvB,
    const float* __restrict__ x, const float* __restrict__ gamma_p,
    float* __restrict__ out)
{
  const int b  = blockIdx.z;
  const int m0 = blockIdx.x * 64;
  const int o0 = blockIdx.y * 64;
  const int tid  = threadIdx.x;
  const int wave = tid >> 5;
  const int lane = tid & 31;
  const int l15  = lane & 15;
  const int half = lane >> 4;
  const int wr = wave >> 1;                    // 0..3 : m sub-tile
  const int wc = wave & 1;                     // 0..1 : o sub-tile pair
  const int mRow  = m0 + wr * 16;
  const int ot0   = (o0 + wc * 32) >> 4;
  const int ot1   = ot0 + 1;

  const float* sarow = sa + ((size_t)b * NN + mRow + l15) * NN;
  const size_t pvbase = (size_t)b * 128 * 16 * 32 * 16;

  v8f acc0 = {0.f,0.f,0.f,0.f,0.f,0.f,0.f,0.f};
  v8f acc1 = {0.f,0.f,0.f,0.f,0.f,0.f,0.f,0.f};

  for (int kt = 0; kt < 128; ++kt) {
    const int kb = kt << 5;
    const float* pa = sarow + kb + 8 * half;
    __builtin_prefetch(sarow + kb + 256, 0, 1);   // global_prefetch_b8
    const float4 a0 = ((const float4*)pa)[0];
    const float4 a1 = ((const float4*)pa)[1];
    const float4 a2 = ((const float4*)(pa + 16))[0];
    const float4 a3 = ((const float4*)(pa + 16))[1];
    v16bf af;
    af[0]=(bf16_t)a0.x;  af[1]=(bf16_t)a0.y;  af[2]=(bf16_t)a0.z;  af[3]=(bf16_t)a0.w;
    af[4]=(bf16_t)a1.x;  af[5]=(bf16_t)a1.y;  af[6]=(bf16_t)a1.z;  af[7]=(bf16_t)a1.w;
    af[8]=(bf16_t)a2.x;  af[9]=(bf16_t)a2.y;  af[10]=(bf16_t)a2.z; af[11]=(bf16_t)a2.w;
    af[12]=(bf16_t)a3.x; af[13]=(bf16_t)a3.y; af[14]=(bf16_t)a3.z; af[15]=(bf16_t)a3.w;

    const v16bf b0 = *(const v16bf*)(pvB + pvbase + (((size_t)kt * 16 + ot0) * 32 + lane) * 16);
    const v16bf b1 = *(const v16bf*)(pvB + pvbase + (((size_t)kt * 16 + ot1) * 32 + lane) * 16);
    acc0 = wmma_bf16(af, b0, acc0);
    acc1 = wmma_bf16(af, b1, acc1);
  }

  const float gamma = gamma_p[0];
#pragma unroll
  for (int t = 0; t < 2; ++t) {
    const v8f acc = (t == 0) ? acc0 : acc1;
    const int o = ((t == 0) ? ot0 : ot1) * 16 + l15;
    const size_t base = ((size_t)b * CC + o) * NN + mRow + 8 * half;
    const float4 xv0 = ((const float4*)(x + base))[0];
    const float4 xv1 = ((const float4*)(x + base))[1];
    float4 r0, r1;
    r0.x = fmaf(gamma, acc[0], xv0.x);
    r0.y = fmaf(gamma, acc[1], xv0.y);
    r0.z = fmaf(gamma, acc[2], xv0.z);
    r0.w = fmaf(gamma, acc[3], xv0.w);
    r1.x = fmaf(gamma, acc[4], xv1.x);
    r1.y = fmaf(gamma, acc[5], xv1.y);
    r1.z = fmaf(gamma, acc[6], xv1.z);
    r1.w = fmaf(gamma, acc[7], xv1.w);
    ((float4*)(out + base))[0] = r0;
    ((float4*)(out + base))[1] = r1;
  }
}

// ---------------------------------------------------------------------------
extern "C" void kernel_launch(void* const* d_in, const int* in_sizes, int n_in,
                              void* d_out, int out_size, void* d_ws, size_t ws_size,
                              hipStream_t stream)
{
  (void)in_sizes; (void)n_in; (void)out_size; (void)ws_size;
  const float* x     = (const float*)d_in[0];
  const float* valid = (const float*)d_in[1];
  const float* Wq    = (const float*)d_in[2];
  const float* bq    = (const float*)d_in[3];
  const float* Wk    = (const float*)d_in[4];
  const float* bk    = (const float*)d_in[5];
  const float* Wv    = (const float*)d_in[6];
  const float* bv    = (const float*)d_in[7];
  const float* gamma = (const float*)d_in[8];

  // workspace: qA (1MB) | kB (1MB) | pvB (8MB), all bf16 fragment-swizzled
  ushort_t* qA  = (ushort_t*)d_ws;
  ushort_t* kB  = qA + (size_t)NB * 256 * 32 * 16;
  ushort_t* pvB = kB + (size_t)NB * 256 * 32 * 16;

  float* outp = (float*)d_out;                       // [B,C,N] fp32
  float* sap  = outp + (size_t)NB * CC * NN;         // [B,N,N] fp32

  proj_qk_kernel    <<<dim3(NN / 256, NB),        256, 0, stream>>>(x, Wq, bq, Wk, bk, qA, kB);
  proj_v_kernel     <<<dim3(NN / 16,  NB),        256, 0, stream>>>(x, Wv, bv, pvB);
  attn_softmax_kernel<<<dim3(NN / 16, NB),        256, 0, stream>>>(qA, kB, valid, sap);
  attn_out_kernel   <<<dim3(NN / 64, CC / 64, NB),256, 0, stream>>>(sap, pvB, x, gamma, outp);
}
